// RNN_31095563223696
// MI455X (gfx1250) — matrix operand order repaired and checked by reading, compiled, not run on hardware
//
#include <hip/hip_runtime.h>
#include <hip/hip_bf16.h>
#include <math.h>

// ---------------------------------------------------------------------------
// 2-layer tanh RNN + decoder for MI455X (gfx1250, wave32, WMMA).
//   - All GEMMs use v_wmma_f32_16x16x32_bf16 (bf16 in, f32 accumulate).
//   - Recurrence: persistent kernel, 64 blocks x 128 thr (256 waves = 4x64
//     tiling of the [64,1024] step output), atomic grid barrier per step,
//     double-buffered bf16 hidden state.
//   - W_hh slice (64 rows = 128KB bf16) is preloaded into LDS ONCE per block
//     (loop-invariant over all 512 steps); the 16-row h slice (32KB) is
//     staged into LDS each step. 160KB/WGP of the 320KB LDS -> 1 block/WGP.
//     This removes ~16MB/step of L2 traffic from the sequential critical path.
// ---------------------------------------------------------------------------

typedef __bf16 bf16;
typedef __attribute__((ext_vector_type(16))) __bf16 v16bf;
typedef __attribute__((ext_vector_type(8)))  __bf16 v8bf;
typedef __attribute__((ext_vector_type(8)))  float  v8f;

constexpr int T_SEQ = 512;
constexpr int BATCH = 64;
constexpr int C_IN  = 256;
constexpr int H_DIM = 1024;

constexpr size_t SCAN_LDS_BYTES = (size_t)(64 + 16) * H_DIM * sizeof(bf16); // 160KB

// ---- WMMA fragment loaders (wave32; A and B share one contiguous k order) -----------
// A: 16x32 bf16 from row-major [.,K] storage.
//   lane l: row = l&15, K-offsets (l>>4)*16 + {0..15} (contiguous).
template <typename PtrT>
__device__ __forceinline__ v16bf load_a_frag(PtrT A, int K, int m, int k, int lane) {
  const int ka = (lane >> 4) * 16;
  auto base = A + (size_t)m * K + k + ka;
  v8bf lo = *(const v8bf*)(base);
  v8bf hi = *(const v8bf*)(base + 8);
  v16bf r;
#pragma unroll
  for (int i = 0; i < 8; ++i) { r[i] = lo[i]; r[8 + i] = hi[i]; }
  return r;
}

// B: 32x16 bf16 where B[k][n] = W[n][k], weights row-major [N,K].
//   lane l: col = l&15, K-offsets (l>>4)*16 + {0..15} (contiguous 32B).
template <typename PtrT>
__device__ __forceinline__ v16bf load_b_frag(PtrT W, int K, int n, int k, int lane) {
  const int kb = (lane >> 4) * 16;
  return *(const v16bf*)(W + (size_t)n * K + k + kb);
}

// ---- elementwise helpers ---------------------------------------------------
__global__ void f32_to_bf16_kernel(const float* __restrict__ in,
                                   bf16* __restrict__ out, int n) {
  int i = blockIdx.x * blockDim.x + threadIdx.x;
  int stride = gridDim.x * blockDim.x;
  for (; i < n; i += stride) out[i] = (bf16)in[i];
}

__global__ void add_bias_kernel(const float* __restrict__ a,
                                const float* __restrict__ b,
                                float* __restrict__ out, int n) {
  int i = blockIdx.x * blockDim.x + threadIdx.x;
  if (i < n) out[i] = a[i] + b[i];
}

// ---- batched GEMM: Out[m,n] = bias[n] + sum_k A[m,k] * W[n,k] --------------
// One wave computes a 16x64 strip (A fragment reused across 4 N-tiles).
__global__ void __launch_bounds__(256)
gemm_bf16_wmma(const bf16* __restrict__ A, const bf16* __restrict__ W,
               const float* __restrict__ bias, float* __restrict__ Out,
               int M, int N, int K) {
  const int lane = threadIdx.x & 31;
  const int wave = blockIdx.x * (blockDim.x >> 5) + (threadIdx.x >> 5);
  const int nChunks = N >> 6;                 // 64-column chunks
  const int m0 = (wave / nChunks) << 4;
  const int n0 = (wave % nChunks) << 6;
  if (m0 >= M) return;

  v8f acc[4] = {};
  const int m  = m0 + (lane & 15);
  const int cl = lane & 15;

  for (int k = 0; k < K; k += 32) {
    v16bf a = load_a_frag(A, K, m, k, lane);
#pragma unroll
    for (int t = 0; t < 4; ++t) {
      v16bf b = load_b_frag(W, K, n0 + (t << 4) + cl, k, lane);
      acc[t] = __builtin_amdgcn_wmma_f32_16x16x32_bf16(
          false, a, false, b, (short)0, acc[t], false, false);
    }
  }

  const int rbase = m0 + ((lane >> 4) << 3);
#pragma unroll
  for (int t = 0; t < 4; ++t) {
    const int col = n0 + (t << 4) + cl;
    const float bv = bias ? bias[col] : 0.0f;
#pragma unroll
    for (int v = 0; v < 8; ++v)
      Out[(size_t)(rbase + v) * N + col] = acc[t][v] + bv;
  }
}

// ---- persistent recurrence: h_t = tanh(XW[t] + h_{t-1} W_hh^T) -------------
// Grid MUST be 64 blocks x 128 threads. Block b: batch tile m0=(b>>4)*16,
// hidden columns n_base=(b&15)*64 (4 waves x one 16x16 tile each).
// Dynamic LDS: [64][1024] bf16 W_hh slice + [16][1024] bf16 h slice.
__global__ void __launch_bounds__(128)
rnn_scan_wmma(const float* __restrict__ XW,    // [T,B,H] f32 (input proj + biases)
              const bf16*  __restrict__ Whh,   // [H,H] bf16 row-major
              bf16* __restrict__ hbuf0,        // [B,H] bf16 (holds h_{-1} at entry)
              bf16* __restrict__ hbuf1,        // [B,H] bf16 double buffer
              bf16* __restrict__ Y,            // [T,B,H] bf16 layer output
              float* __restrict__ hlast,       // [B,H] f32 final hidden (d_out)
              unsigned* __restrict__ bar) {
  extern __shared__ char smem[];
  bf16* Bsh = (bf16*)smem;                               // [64][H] W_hh slice
  bf16* Ash = (bf16*)(smem + (size_t)64 * H_DIM * 2);    // [16][H] h slice

  const int tid    = threadIdx.x;
  const int lane   = tid & 31;
  const int w      = tid >> 5;                 // wave in block, 0..3
  const int m0     = (blockIdx.x >> 4) << 4;   // batch tile base (global)
  const int n_base = (blockIdx.x & 15) << 6;   // block's 64 hidden columns
  const int n_loc  = w << 4;                   // this wave's 16 cols (LDS-local)
  const int cl     = lane & 15;
  const int rloc   = (lane >> 4) << 3;         // 0 or 8 (local row base)
  unsigned* cnt = bar;
  unsigned* gen = bar + 16;

  // ---- preload loop-invariant W_hh slice (contiguous 128KB) into LDS ----
  {
    const v8bf* src = (const v8bf*)(Whh + (size_t)n_base * H_DIM);
    v8bf* dst = (v8bf*)Bsh;
    for (int i = tid; i < (64 * H_DIM) / 8; i += 128) dst[i] = src[i];
  }
  __syncthreads();

  for (int t = 0; t < T_SEQ; ++t) {
    const bf16* hin  = (t & 1) ? hbuf1 : hbuf0;
    bf16*       hnew = (t & 1) ? hbuf0 : hbuf1;

    // ---- stage this block's 16 h rows (contiguous 32KB) into LDS ----
    {
      const v8bf* src = (const v8bf*)(hin + (size_t)m0 * H_DIM);
      v8bf* dst = (v8bf*)Ash;
      for (int i = tid; i < (16 * H_DIM) / 8; i += 128) dst[i] = src[i];
    }
    __syncthreads();

    v8f acc = {};
    for (int k = 0; k < H_DIM; k += 32) {
      v16bf a = load_a_frag(Ash, H_DIM, cl, k, lane);            // LDS
      v16bf b = load_b_frag(Bsh, H_DIM, n_loc + cl, k, lane);    // LDS
      acc = __builtin_amdgcn_wmma_f32_16x16x32_bf16(
          false, a, false, b, (short)0, acc, false, false);
    }

    const int    col = n_base + n_loc + cl;     // global hidden column
    const float* xw  = XW + (size_t)t * BATCH * H_DIM;
    bf16*        yt  = Y  + (size_t)t * BATCH * H_DIM;
#pragma unroll
    for (int v = 0; v < 8; ++v) {
      const size_t idx = (size_t)(m0 + rloc + v) * H_DIM + col;
      float val = tanhf(acc[v] + xw[idx]);
      bf16 hv = (bf16)val;
      hnew[idx] = hv;
      yt[idx]   = hv;
      if (t == T_SEQ - 1) hlast[idx] = val;
    }

    // ---- grid-wide barrier (all 64 blocks co-resident, 1 per WGP) ----
    __syncthreads();
    if (tid == 0) {
      __threadfence();  // make h/Y writes device-visible
      unsigned g = __hip_atomic_load(gen, __ATOMIC_ACQUIRE,
                                     __HIP_MEMORY_SCOPE_AGENT);
      unsigned arrived = __hip_atomic_fetch_add(cnt, 1u, __ATOMIC_ACQ_REL,
                                                __HIP_MEMORY_SCOPE_AGENT);
      if (arrived == gridDim.x - 1) {
        __hip_atomic_store(cnt, 0u, __ATOMIC_RELAXED,
                           __HIP_MEMORY_SCOPE_AGENT);
        __hip_atomic_fetch_add(gen, 1u, __ATOMIC_ACQ_REL,
                               __HIP_MEMORY_SCOPE_AGENT);
      } else {
        while (__hip_atomic_load(gen, __ATOMIC_ACQUIRE,
                                 __HIP_MEMORY_SCOPE_AGENT) == g) {
          __builtin_amdgcn_s_sleep(2);
        }
      }
    }
    __syncthreads();
  }
}

// ---------------------------------------------------------------------------
extern "C" void kernel_launch(void* const* d_in, const int* in_sizes, int n_in,
                              void* d_out, int out_size, void* d_ws, size_t ws_size,
                              hipStream_t stream) {
  const float* x     = (const float*)d_in[0];
  const float* h0    = (const float*)d_in[1];
  const float* w_ih0 = (const float*)d_in[2];
  const float* w_hh0 = (const float*)d_in[3];
  const float* b_ih0 = (const float*)d_in[4];
  const float* b_hh0 = (const float*)d_in[5];
  const float* w_ih1 = (const float*)d_in[6];
  const float* w_hh1 = (const float*)d_in[7];
  const float* b_ih1 = (const float*)d_in[8];
  const float* b_hh1 = (const float*)d_in[9];
  const float* w_dec = (const float*)d_in[10];
  const float* b_dec = (const float*)d_in[11];
  float* out = (float*)d_out;   // [T*B*C decoded][2*B*H hidden]

  // allow 160KB dynamic LDS for the scan kernel (host-side, capture-safe)
  static bool lds_cfg = false;
  if (!lds_cfg) {
    hipFuncSetAttribute((const void*)rnn_scan_wmma,
                        hipFuncAttributeMaxDynamicSharedMemorySize,
                        (int)SCAN_LDS_BYTES);
    lds_cfg = true;
  }

  // ---- workspace carve-up (256B aligned) ----
  char* ws = (char*)d_ws;
  size_t off = 0;
  auto carve = [&](size_t bytes) -> void* {
    void* p = ws + off;
    off = (off + bytes + 255) & ~(size_t)255;
    return p;
  };
  unsigned* bar   = (unsigned*)carve(256);
  float* bias0    = (float*)carve(H_DIM * 4);
  float* bias1    = (float*)carve(H_DIM * 4);
  bf16* x_bf      = (bf16*)carve((size_t)T_SEQ * BATCH * C_IN * 2);
  bf16* wih0_bf   = (bf16*)carve((size_t)H_DIM * C_IN * 2);
  bf16* whh0_bf   = (bf16*)carve((size_t)H_DIM * H_DIM * 2);
  bf16* wih1_bf   = (bf16*)carve((size_t)H_DIM * H_DIM * 2);
  bf16* whh1_bf   = (bf16*)carve((size_t)H_DIM * H_DIM * 2);
  bf16* wdec_bf   = (bf16*)carve((size_t)C_IN * H_DIM * 2);
  float* XW       = (float*)carve((size_t)T_SEQ * BATCH * H_DIM * 4);
  bf16* y0_bf     = (bf16*)carve((size_t)T_SEQ * BATCH * H_DIM * 2);
  bf16* y1_bf     = (bf16*)carve((size_t)T_SEQ * BATCH * H_DIM * 2);
  bf16* hA        = (bf16*)carve((size_t)BATCH * H_DIM * 2);
  bf16* hB        = (bf16*)carve((size_t)BATCH * H_DIM * 2);

  hipMemsetAsync(bar, 0, 256, stream);  // barrier counters (graph-safe)

  auto cvt = [&](const float* src, bf16* dst, int n) {
    int blocks = (n + 255) / 256;
    if (blocks > 2048) blocks = 2048;
    f32_to_bf16_kernel<<<blocks, 256, 0, stream>>>(src, dst, n);
  };
  cvt(x,     x_bf,    T_SEQ * BATCH * C_IN);
  cvt(w_ih0, wih0_bf, H_DIM * C_IN);
  cvt(w_hh0, whh0_bf, H_DIM * H_DIM);
  cvt(w_ih1, wih1_bf, H_DIM * H_DIM);
  cvt(w_hh1, whh1_bf, H_DIM * H_DIM);
  cvt(w_dec, wdec_bf, C_IN * H_DIM);
  add_bias_kernel<<<(H_DIM + 255) / 256, 256, 0, stream>>>(b_ih0, b_hh0, bias0, H_DIM);
  add_bias_kernel<<<(H_DIM + 255) / 256, 256, 0, stream>>>(b_ih1, b_hh1, bias1, H_DIM);

  const int M = T_SEQ * BATCH;  // 32768
  auto gemm = [&](const bf16* A, const bf16* W, const float* bias, float* Out,
                  int m, int n, int k) {
    int waves = (m / 16) * (n / 64);
    int blocks = (waves + 7) / 8;   // 8 waves per 256-thread block
    gemm_bf16_wmma<<<blocks, 256, 0, stream>>>(A, W, bias, Out, m, n, k);
  };

  float* hidden_out = out + (size_t)T_SEQ * BATCH * C_IN;

  // ---- layer 0 ----
  gemm(x_bf, wih0_bf, bias0, XW, M, H_DIM, C_IN);         // XW0 = x W_ih0^T + b
  cvt(h0, hA, BATCH * H_DIM);                             // h_{-1} (layer 0)
  rnn_scan_wmma<<<64, 128, SCAN_LDS_BYTES, stream>>>(XW, whh0_bf, hA, hB, y0_bf,
                                                     hidden_out, bar);

  // ---- layer 1 ----
  gemm(y0_bf, wih1_bf, bias1, XW, M, H_DIM, H_DIM);       // XW1 = y0 W_ih1^T + b
  cvt(h0 + BATCH * H_DIM, hA, BATCH * H_DIM);             // h_{-1} (layer 1)
  rnn_scan_wmma<<<64, 128, SCAN_LDS_BYTES, stream>>>(XW, whh1_bf, hA, hB, y1_bf,
                                                     hidden_out + BATCH * H_DIM, bar);

  // ---- decoder ----
  gemm(y1_bf, wdec_bf, b_dec, out, M, C_IN, H_DIM);       // decoded = y1 W_dec^T + b
}
